// TGCN_43301860278673
// MI455X (gfx1250) — matrix operand order repaired and checked by reading, not compiled, not run on hardware
//
#include <hip/hip_runtime.h>
#include <hip/hip_bf16.h>
#include <cstdint>

// ---- problem constants ----
#define BB 128
#define SS 12
#define NN 325
#define DD 64
#define BSZ 1536          // BB*SS
#define NP 336            // NN padded to 21*16
#define KP 352            // K padded to 11*32
#define MT 21             // 16-row tiles covering NN..NP
#define MT2 22            // 16-row tiles covering KP rows (for zero padding)

typedef __attribute__((ext_vector_type(16))) _Float16 v16h;
typedef __attribute__((ext_vector_type(8)))  _Float16 v8h;
typedef __attribute__((ext_vector_type(8)))  float    v8f;

// ---------------------------------------------------------------------------
// WMMA helpers (CDNA5 wave32, V_WMMA_F32_16X16X32_F16)
// A frag (16x32 f16, MxK): lanes 0-15 -> M=lane, halves j<8: K=kb+j, j>=8: K=16+kb+j
//                          (kb = 8 for lanes 16-31), per ISA 7.12.2 table.
// B frag (32x16 f16, KxN): lanes&15 -> N col, halves j<8: K=kb+j, j>=8: K=16+kb+j (kb=8 for hi lanes).
// C/D frag (16x16 f32):    vgpr r -> M = rowb + r (rowb=8 for hi lanes), N = lane&15.
// ---------------------------------------------------------------------------
__device__ __forceinline__ v8f wmma32(v16h a, v16h b, v8f c) {
  return __builtin_amdgcn_wmma_f32_16x16x32_f16(false, a, false, b, (short)0, c,
                                                false, false);
}

// A fragment from a row-major f16 tile (tile points at row 0 of the 16-row tile)
__device__ __forceinline__ v16h load_a16(const _Float16* tile, int pitch, int k0, int lane) {
  const _Float16* p = tile + (size_t)(lane & 15) * pitch + k0 + ((lane & 16) ? 8 : 0);
  v8h lo = *(const v8h*)p;
  v8h hi = *(const v8h*)(p + 16);
  v16h r;
#pragma unroll
  for (int j = 0; j < 8; ++j) { r[j] = lo[j]; r[8 + j] = hi[j]; }
  return r;
}

// B fragment from per-column-contiguous f16 storage: element (k, col) at col*pitch + k
__device__ __forceinline__ v16h load_b16(const _Float16* colmaj, int pitch, int k0, int lane) {
  const _Float16* p = colmaj + (size_t)(lane & 15) * pitch + k0 + ((lane & 16) ? 8 : 0);
  v8h lo = *(const v8h*)p;
  v8h hi = *(const v8h*)(p + 16);
  v16h r;
#pragma unroll
  for (int j = 0; j < 8; ++j) { r[j] = lo[j]; r[8 + j] = hi[j]; }
  return r;
}

__device__ __forceinline__ float sigmf(float x) { return 1.f / (1.f + __expf(-x)); }

// ---------------------------------------------------------------------------
// Prep: f16 conversions into WMMA-friendly layouts
// ---------------------------------------------------------------------------
__global__ void k_prep_adj(const float* __restrict__ adj, _Float16* __restrict__ adjH) {
  int idx = blockIdx.x * blockDim.x + threadIdx.x;
  if (idx >= NP * KP) return;
  int m = idx / KP, n = idx % KP;
  float v = (m < NN && n < NN) ? adj[m * NN + n] : 0.f;
  adjH[idx] = (_Float16)v;
}

__global__ void k_prep_x(const float* __restrict__ x, _Float16* __restrict__ XH) {
  int idx = blockIdx.x * blockDim.x + threadIdx.x;
  if (idx >= BSZ * KP) return;
  int bs = idx / KP, n = idx % KP;
  float v = (n < NN) ? x[(size_t)bs * NN + n] : 0.f;
  XH[idx] = (_Float16)v;   // XH[bs][n]  (per-column-contiguous B storage for GEMM1)
}

__global__ void k_prep_w(const float* __restrict__ Wih, const float* __restrict__ Whh,
                         const float* __restrict__ W2,
                         _Float16* __restrict__ WihH, _Float16* __restrict__ WhhH,
                         _Float16* __restrict__ W2T) {
  int idx = blockIdx.x * blockDim.x + threadIdx.x;
  if (idx < 3 * DD * DD) {                       // Wih [192][64]: rows = out col -> already col-contig
    WihH[idx] = (_Float16)Wih[idx];
  } else if (idx < 6 * DD * DD) {
    int j = idx - 3 * DD * DD;
    WhhH[j] = (_Float16)Whh[j];
  } else if (idx < 6 * DD * DD + DD * DD) {      // W2T[e][k] = W2[k][e]
    int j = idx - 6 * DD * DD;
    int e = j >> 6, k = j & 63;
    W2T[e * DD + k] = (_Float16)W2[k * DD + e];
  }
}

// ---------------------------------------------------------------------------
// K1: Y1[m, bs] = sum_n adj[m,n] * x[bs,n]    (M=336, N=1536, K=352)
// one wave per 16x16 tile
// ---------------------------------------------------------------------------
__global__ void k_gemm1(const _Float16* __restrict__ adjH, const _Float16* __restrict__ XH,
                        float* __restrict__ Y1) {
  int mt = blockIdx.x;                 // 0..20
  int bt = blockIdx.y;                 // 0..95
  int lane = threadIdx.x;              // 32 threads
  const _Float16* arow = adjH + (size_t)(mt * 16) * KP;
  const _Float16* bcol = XH + (size_t)(bt * 16) * KP;
  v8f acc = {};
#pragma unroll
  for (int ks = 0; ks < KP; ks += 32) {
    v16h a = load_a16(arow, KP, ks, lane);
    v16h b = load_b16(bcol, KP, ks, lane);
    acc = wmma32(a, b, acc);
  }
  int col = bt * 16 + (lane & 15);
  int rowb = mt * 16 + ((lane & 16) ? 8 : 0);
#pragma unroll
  for (int r = 0; r < 8; ++r) Y1[(size_t)(rowb + r) * BSZ + col] = acc[r];
}

// ---------------------------------------------------------------------------
// K2: Tp[bs][e][n] = sum_d relu(Y1[n,bs]*W1[d]+b1[d]) * W2[d,e]
// block = 4 waves; wave w -> e-tile 16w. Covers n rows 0..351 (zeros for n>=325).
// ---------------------------------------------------------------------------
__global__ void k_gcn2a(const float* __restrict__ Y1, const float* __restrict__ W1,
                        const float* __restrict__ b1, const _Float16* __restrict__ W2T,
                        _Float16* __restrict__ Tp) {
  int mt = blockIdx.x;                 // 0..21
  int bs = blockIdx.y;                 // 0..1535
  int wave = threadIdx.x >> 5;
  int lane = threadIdx.x & 31;
  int rowm = mt * 16 + (lane & 15);
  bool valid = rowm < NN;
  float yv = valid ? Y1[(size_t)rowm * BSZ + bs] : 0.f;
  int kb = (lane & 16) ? 8 : 0;
  v16h a0, a1;
#pragma unroll
  for (int j = 0; j < 8; ++j) {
    int dA = kb + j, dB = 16 + kb + j, dC = 32 + kb + j, dE = 48 + kb + j;
    a0[j]     = valid ? (_Float16)fmaxf(yv * W1[dA] + b1[dA], 0.f) : (_Float16)0.f;
    a0[8 + j] = valid ? (_Float16)fmaxf(yv * W1[dB] + b1[dB], 0.f) : (_Float16)0.f;
    a1[j]     = valid ? (_Float16)fmaxf(yv * W1[dC] + b1[dC], 0.f) : (_Float16)0.f;
    a1[8 + j] = valid ? (_Float16)fmaxf(yv * W1[dE] + b1[dE], 0.f) : (_Float16)0.f;
  }
  int e0 = wave * 16;
  v16h b0 = load_b16(W2T + (size_t)e0 * DD, DD, 0, lane);
  v16h b1f = load_b16(W2T + (size_t)e0 * DD, DD, 32, lane);
  v8f acc = {};
  acc = wmma32(a0, b0, acc);
  acc = wmma32(a1, b1f, acc);
  // store D tile to Tp[bs][e][m]: per lane 8 contiguous m values
  int colE = e0 + (lane & 15);
  int rowb = mt * 16 + ((lane & 16) ? 8 : 0);
  v8h st;
#pragma unroll
  for (int r = 0; r < 8; ++r) st[r] = (_Float16)acc[r];
  *(v8h*)(Tp + ((size_t)bs * DD + colE) * KP + rowb) = st;
}

// ---------------------------------------------------------------------------
// K3: H2[bs][m][e] = relu( sum_n adj[m,n]*Tp[bs][e][n] + b2[e] )
// 1536 batched 325x64x352 GEMMs. Dominant op (~21 GFLOP).
// The 16x352 f16 adjacency A-tile is shared by all 4 waves of the block:
// stage it once into LDS with per-lane GLOBAL_LOAD_ASYNC_TO_LDS_B128
// (ASYNCcnt-tracked, decoupled from VALU), then read A-fragments via ds loads.
// ---------------------------------------------------------------------------
__global__ void k_gcn2b(const _Float16* __restrict__ adjH, const _Float16* __restrict__ Tp,
                        const float* __restrict__ b2, _Float16* __restrict__ H2) {
  __shared__ __align__(16) _Float16 Atile[16][KP];   // 11264 bytes, contiguous
  int mt = blockIdx.x;                 // 0..20
  int bs = blockIdx.y;                 // 0..1535
  int tid = threadIdx.x;
  int wave = tid >> 5;
  int lane = tid & 31;

  // --- async stage A tile: flat 704 x 16B copy global -> LDS ---
  {
    const char* gsrc = (const char*)(adjH + (size_t)(mt * 16) * KP);
    unsigned ldsbase = (unsigned)(uintptr_t)(&Atile[0][0]);  // low 32 bits = LDS offset
    for (int i = tid; i < (16 * KP * 2) / 16; i += 128) {
      uint64_t g = (uint64_t)(uintptr_t)(gsrc + i * 16);
      unsigned l = ldsbase + i * 16;
      asm volatile("global_load_async_to_lds_b128 %0, %1, off"
                   :: "v"(l), "v"(g) : "memory");
    }
    asm volatile("s_wait_asynccnt 0x0" ::: "memory");
  }
  __syncthreads();

  int e0 = wave * 16;
  const _Float16* bcol = Tp + ((size_t)bs * DD + e0) * KP;
  const _Float16* ap = &Atile[lane & 15][0];
  int kb = (lane & 16) ? 8 : 0;
  v8f acc = {};
#pragma unroll
  for (int ks = 0; ks < KP; ks += 32) {
    __builtin_prefetch(bcol + (size_t)(lane & 15) * KP + ks + 64, 0, 1); // global_prefetch_b8
    v8h lo = *(const v8h*)(ap + ks + kb);
    v8h hi = *(const v8h*)(ap + ks + kb + 16);
    v16h a;
#pragma unroll
    for (int j = 0; j < 8; ++j) { a[j] = lo[j]; a[8 + j] = hi[j]; }
    v16h b = load_b16(bcol, KP, ks, lane);
    acc = wmma32(a, b, acc);
  }
  int colE = e0 + (lane & 15);
  int rowb = (lane & 16) ? 8 : 0;
  float bias = b2[colE];
#pragma unroll
  for (int r = 0; r < 8; ++r) {
    int m = mt * 16 + rowb + r;
    if (m < NN)
      H2[((size_t)bs * NN + m) * DD + colE] = (_Float16)fmaxf(acc[r] + bias, 0.f);
  }
}

// ---------------------------------------------------------------------------
// K4: fused 12-step GRU + output head. One block = 16 rows (b, m-tile); 4 waves
// each own 16 hidden columns. Hidden state exchanged via LDS each step
// (C-layout -> A-layout transpose). Weight B-fragments stay in registers.
// ---------------------------------------------------------------------------
__global__ void k_gru(const _Float16* __restrict__ H2, const _Float16* __restrict__ WihH,
                      const _Float16* __restrict__ WhhH, const float* __restrict__ bih,
                      const float* __restrict__ bhh, const float* __restrict__ Wout,
                      const float* __restrict__ bout, float* __restrict__ out) {
  __shared__ __align__(32) _Float16 hs[16][72];   // 16 rows x 64 d (+pad)
  int mt = blockIdx.x;                 // 0..20
  int b  = blockIdx.y;                 // 0..127
  int tid = threadIdx.x;
  int wave = tid >> 5;
  int lane = tid & 31;
  int hc0 = wave * 16;
  int colh = hc0 + (lane & 15);

  for (int i = tid; i < 16 * 72; i += blockDim.x) (&hs[0][0])[i] = (_Float16)0.f;
  __syncthreads();

  // loop-invariant weight B fragments: 3 gates x 2 k-steps, for Wih and Whh
  v16h Bi[3][2], Bh[3][2];
#pragma unroll
  for (int g = 0; g < 3; ++g) {
    const _Float16* wi = WihH + (size_t)(g * DD + hc0) * DD;
    const _Float16* wh = WhhH + (size_t)(g * DD + hc0) * DD;
    Bi[g][0] = load_b16(wi, DD, 0, lane);
    Bi[g][1] = load_b16(wi, DD, 32, lane);
    Bh[g][0] = load_b16(wh, DD, 0, lane);
    Bh[g][1] = load_b16(wh, DD, 32, lane);
  }
  float bi[3], bh[3];
#pragma unroll
  for (int g = 0; g < 3; ++g) { bi[g] = bih[g * DD + colh]; bh[g] = bhh[g * DD + colh]; }

  int rowm = mt * 16 + (lane & 15);
  bool valid = rowm < NN;
  int kb = (lane & 16) ? 8 : 0;
  int rl = lane & 15;
  v8f hprev = {};

  for (int s = 0; s < SS; ++s) {
    // A fragments of x_t from H2 (row-major [.,64] f16)
    v16h ax0, ax1;
    if (valid) {
      const _Float16* xb = H2 + ((size_t)(b * SS + s) * NN + rowm) * DD;
      v8h l0 = *(const v8h*)(xb + kb);
      v8h h0 = *(const v8h*)(xb + 16 + kb);
      v8h l1 = *(const v8h*)(xb + 32 + kb);
      v8h h1 = *(const v8h*)(xb + 48 + kb);
#pragma unroll
      for (int j = 0; j < 8; ++j) {
        ax0[j] = l0[j]; ax0[8 + j] = h0[j];
        ax1[j] = l1[j]; ax1[8 + j] = h1[j];
      }
    } else {
#pragma unroll
      for (int j = 0; j < 16; ++j) { ax0[j] = (_Float16)0.f; ax1[j] = (_Float16)0.f; }
    }
    // A fragments of h_{t-1} from LDS
    v16h ah0, ah1;
    {
      v8h l0 = *(const v8h*)(&hs[rl][kb]);
      v8h h0 = *(const v8h*)(&hs[rl][16 + kb]);
      v8h l1 = *(const v8h*)(&hs[rl][32 + kb]);
      v8h h1 = *(const v8h*)(&hs[rl][48 + kb]);
#pragma unroll
      for (int j = 0; j < 8; ++j) {
        ah0[j] = l0[j]; ah0[8 + j] = h0[j];
        ah1[j] = l1[j]; ah1[8 + j] = h1[j];
      }
    }
    v8f gi[3], gh[3];
#pragma unroll
    for (int g = 0; g < 3; ++g) {
      v8f z = {};
      z = wmma32(ax0, Bi[g][0], z);
      gi[g] = wmma32(ax1, Bi[g][1], z);
      v8f z2 = {};
      z2 = wmma32(ah0, Bh[g][0], z2);
      gh[g] = wmma32(ah1, Bh[g][1], z2);
    }
    __syncthreads();   // everyone done reading hs before it is overwritten
    v8f hnew;
#pragma unroll
    for (int r = 0; r < 8; ++r) {
      float rr = sigmf(gi[0][r] + bi[0] + gh[0][r] + bh[0]);
      float zz = sigmf(gi[1][r] + bi[1] + gh[1][r] + bh[1]);
      float cc = tanhf(gi[2][r] + bi[2] + rr * (gh[2][r] + bh[2]));
      hnew[r] = (1.f - zz) * cc + zz * hprev[r];
    }
    hprev = hnew;
    int rowb = (lane & 16) ? 8 : 0;
#pragma unroll
    for (int r = 0; r < 8; ++r) hs[rowb + r][colh] = (_Float16)hprev[r];
    __syncthreads();
  }

  // output head: out[b, m] = h . Wout + bout
  if (tid < 16) {
    int m = mt * 16 + tid;
    if (m < NN) {
      float acc = bout[0];
#pragma unroll
      for (int d = 0; d < DD; ++d) acc += (float)hs[tid][d] * Wout[d];
      out[(size_t)b * NN + m] = acc;
    }
  }
}

// ---------------------------------------------------------------------------
extern "C" void kernel_launch(void* const* d_in, const int* in_sizes, int n_in,
                              void* d_out, int out_size, void* d_ws, size_t ws_size,
                              hipStream_t stream) {
  const float* x    = (const float*)d_in[0];
  const float* adj  = (const float*)d_in[1];
  const float* W1   = (const float*)d_in[2];
  const float* b1   = (const float*)d_in[3];
  const float* W2   = (const float*)d_in[4];
  const float* b2   = (const float*)d_in[5];
  const float* Wih  = (const float*)d_in[6];
  const float* Whh  = (const float*)d_in[7];
  const float* bih  = (const float*)d_in[8];
  const float* bhh  = (const float*)d_in[9];
  const float* Wout = (const float*)d_in[10];
  const float* bout = (const float*)d_in[11];
  float* out = (float*)d_out;

  char* ws = (char*)d_ws;
  size_t off = 0;
  auto take = [&](size_t bytes) { size_t o = off; off = (off + bytes + 255) & ~(size_t)255; return o; };
  _Float16* adjH = (_Float16*)(ws + take((size_t)NP * KP * 2));
  _Float16* XH   = (_Float16*)(ws + take((size_t)BSZ * KP * 2));
  _Float16* WihH = (_Float16*)(ws + take((size_t)3 * DD * DD * 2));
  _Float16* WhhH = (_Float16*)(ws + take((size_t)3 * DD * DD * 2));
  _Float16* W2T  = (_Float16*)(ws + take((size_t)DD * DD * 2));
  float*    Y1   = (float*)(ws + take((size_t)NP * BSZ * 4));
  _Float16* Tp   = (_Float16*)(ws + take((size_t)BSZ * DD * KP * 2));
  _Float16* H2   = (_Float16*)(ws + take((size_t)BSZ * NN * DD * 2));

  k_prep_adj<<<(NP * KP + 255) / 256, 256, 0, stream>>>(adj, adjH);
  k_prep_x<<<(BSZ * KP + 255) / 256, 256, 0, stream>>>(x, XH);
  k_prep_w<<<(7 * DD * DD + 255) / 256, 256, 0, stream>>>(Wih, Whh, W2, WihH, WhhH, W2T);

  k_gemm1<<<dim3(MT, BSZ / 16), 32, 0, stream>>>(adjH, XH, Y1);
  k_gcn2a<<<dim3(MT2, BSZ), 128, 0, stream>>>(Y1, W1, b1, W2T, Tp);
  k_gcn2b<<<dim3(MT, BSZ), 128, 0, stream>>>(adjH, Tp, b2, H2);
  k_gru<<<dim3(MT, BB), 128, 0, stream>>>(H2, WihH, WhhH, bih, bhh, Wout, bout, out);
}
